// MultiViewFeatureTransformer_45767171506277
// MI455X (gfx1250) — matrix operand order repaired and checked by reading, not compiled, not run on hardware
//
#include <hip/hip_runtime.h>
#include <cstdint>

typedef __bf16 bf16_t;
typedef __attribute__((ext_vector_type(16))) __bf16 v16bf;
typedef __attribute__((ext_vector_type(8)))  __bf16 v8bf;
typedef __attribute__((ext_vector_type(8)))  float  v8f;

constexpr int kV = 3, kBt = 2, kC = 256, kPX = 32, kPY = 32;
constexpr int kL = kPX * kPY;          // 1024 tokens per image
constexpr int kN = kV * kBt;           // 6
constexpr int kNH = 4, kHD = 64;
constexpr int kFFNin = 2 * kC;         // 512
constexpr int kFFNhid = 4 * kFFNin;    // 2048
constexpr int kKVL = (kV - 1) * kL;    // 2048

// ---------------------------------------------------------------- helpers

__device__ inline v16bf frag_from2(const bf16_t* p0, const bf16_t* p1) {
  v8bf lo = *(const v8bf*)p0;
  v8bf hi = *(const v8bf*)p1;
  v16bf r;
#pragma unroll
  for (int i = 0; i < 8; ++i) { r[i] = lo[i]; r[i + 8] = hi[i]; }
  return r;
}

__device__ inline v8f wmma_bf16(v16bf a, v16bf b, v8f c) {
  return __builtin_amdgcn_wmma_f32_16x16x32_bf16(false, a, false, b, (short)0, c,
                                                 false, false);
}

static inline int cdiv(long a, int b) { return (int)((a + b - 1) / b); }

// ---------------------------------------------------------------- f32 -> bf16

__global__ __launch_bounds__(256) void cvt_bf16_kernel(const float* __restrict__ in,
                                                       bf16_t* __restrict__ out, long n) {
  long i = (long)blockIdx.x * blockDim.x + threadIdx.x;
  if (i < n) out[i] = (bf16_t)in[i];
}

// ---------------------------------------------------------------- PRoPE matrices

__device__ inline void prope_mats(const float* vm, const float* Kp, int l,
                                  float* M, float* Mi) {
  float u  = ((l & 31) + 0.5f) * (1.0f / 32.0f);
  float vv = ((l >> 5) + 0.5f) * (1.0f / 32.0f);
  float fx  = Kp[0] * (1.0f / 128.0f);
  float fy  = Kp[4] * (1.0f / 128.0f);
  float cxu = Kp[2] * (1.0f / 128.0f) - u;
  float cyv = Kp[5] * (1.0f / 128.0f) - vv;
#pragma unroll
  for (int j = 0; j < 4; ++j) {
    M[0 * 4 + j] = fx * vm[0 * 4 + j] + cxu * vm[2 * 4 + j];
    M[1 * 4 + j] = fy * vm[1 * 4 + j] + cyv * vm[2 * 4 + j];
    M[2 * 4 + j] = vm[2 * 4 + j];
    M[3 * 4 + j] = vm[3 * 4 + j];
  }
  if (Mi) {  // M^{-1} = vm^{-1} T^{-1} with rigid vm = [R t; 0 1]
    float a = -cxu / fx, bb = -cyv / fy;
#pragma unroll
    for (int i = 0; i < 3; ++i) {
      float Ri0 = vm[0 * 4 + i], Ri1 = vm[1 * 4 + i], Ri2 = vm[2 * 4 + i];
      float ti = -(Ri0 * vm[3] + Ri1 * vm[7] + Ri2 * vm[11]);
      Mi[i * 4 + 0] = Ri0 / fx;
      Mi[i * 4 + 1] = Ri1 / fy;
      Mi[i * 4 + 2] = Ri0 * a + Ri1 * bb + Ri2;
      Mi[i * 4 + 3] = ti;
    }
    Mi[12] = 0.f; Mi[13] = 0.f; Mi[14] = 0.f; Mi[15] = 1.f;
  }
}

__global__ __launch_bounds__(256) void build_mq_kernel(const float* __restrict__ vm,
    const float* __restrict__ Ks, float* __restrict__ Mq, float* __restrict__ Minv) {
  long idx = (long)blockIdx.x * blockDim.x + threadIdx.x;
  if (idx >= (long)kN * kL) return;
  int l = (int)(idx % kL);
  int n = (int)(idx / kL);
  prope_mats(vm + (long)n * 16, Ks + (long)n * 9, l, Mq + idx * 16, Minv + idx * 16);
}

__global__ __launch_bounds__(256) void build_mkv_kernel(const float* __restrict__ vm,
    const float* __restrict__ Ks, float* __restrict__ Mkv) {
  long idx = (long)blockIdx.x * blockDim.x + threadIdx.x;
  if (idx >= (long)kN * kKVL) return;
  int t = (int)(idx % kKVL);
  int n = (int)(idx / kKVL);
  int v = n / kBt, b = n % kBt;
  int m = t / kL, l = t % kL;
  int vs = (m < v) ? m : m + 1;          // IDX[v][m]
  int src = vs * kBt + b;
  prope_mats(vm + (long)src * 16, Ks + (long)src * 9, l, Mkv + idx * 16, nullptr);
}

// ---------------------------------------------------------------- layout shuffles

__global__ __launch_bounds__(256) void feat2x_kernel(const float* __restrict__ feats,
    float* __restrict__ x, bf16_t* __restrict__ xh) {
  long idx = (long)blockIdx.x * blockDim.x + threadIdx.x;
  if (idx >= (long)kN * kL * kC) return;
  int c = (int)(idx % kC);
  long r = idx / kC;
  int l = (int)(r % kL);
  int n = (int)(r / kL);
  float v = feats[((long)n * kC + c) * kL + l];
  x[idx] = v;
  xh[idx] = (bf16_t)v;
}

__global__ __launch_bounds__(256) void gather_kv_kernel(const bf16_t* __restrict__ xh,
    bf16_t* __restrict__ kvxh) {
  long idx = (long)blockIdx.x * blockDim.x + threadIdx.x;
  if (idx >= (long)kN * kKVL * kC) return;
  int c = (int)(idx % kC);
  long r = idx / kC;
  int t = (int)(r % kKVL);
  int n = (int)(r / kKVL);
  int v = n / kBt, b = n % kBt;
  int m = t / kL, l = t % kL;
  int vs = (m < v) ? m : m + 1;
  int nsrc = vs * kBt + b;
  kvxh[idx] = xh[((long)nsrc * kL + l) * kC + c];
}

__global__ __launch_bounds__(256) void out_kernel(const float* __restrict__ x,
                                                  float* __restrict__ out) {
  long idx = (long)blockIdx.x * blockDim.x + threadIdx.x;
  if (idx >= (long)kN * kC * kL) return;
  int l = (int)(idx % kL);
  long r = idx / kL;
  int c = (int)(r % kC);
  int n = (int)(r / kC);
  out[idx] = x[((long)n * kL + l) * kC + c];
}

// ---------------------------------------------------------------- generic NT GEMM
// C[b][m][n] = sum_k A[b][m][k] * Bw[n][k]  (Bw = weights, shared across batch)
// Block = 128(M) x 64(N) super-tile, 8 waves x (16x64). The 64-row weight tile
// is staged through LDS with double-buffered async global->LDS copies
// (ASYNCcnt path), shared by all 8 waves; each wave reuses its A fragment
// across 4 WMMAs. Grids are exact, so the barriers are uniform.

__global__ __launch_bounds__(256) void gemm_nt_kernel(const bf16_t* __restrict__ A,
    const bf16_t* __restrict__ Bw, float* __restrict__ Cc,
    int M, int Nn, int K, long strideA, long strideC) {
  __shared__ alignas(16) bf16_t Blds[2][64 * 64];
  int lane = threadIdx.x & 31;
  int wave = threadIdx.x >> 5;
  int hl = lane >> 4;
  int lid = lane & 15;
  int nB = Nn >> 6;        // N super-tiles
  int mB = M >> 7;         // M super-tiles
  int t = blockIdx.x;
  int jnS = t % nB; t /= nB;
  int imS = t % mB;
  int b = t / mB;

  const bf16_t* Arow = A + (long)b * strideA + (long)(imS * 128 + wave * 16 + lid) * K;
  const bf16_t* Bbase = Bw + (long)(jnS * 64) * K;

  // async-stage one 64x64 bf16 chunk of the weight tile into LDS buffer `bufsel`
  auto stage = [&](int kk, int bufsel) {
#pragma unroll
    for (int it = 0; it < 2; ++it) {
      int s = (int)threadIdx.x + it * 256;       // 512 x 16B segments
      int row = s >> 3;
      int col = (s & 7) * 8;
      const bf16_t* g = Bbase + (long)row * K + kk + col;
      unsigned ldsoff = (unsigned)(uintptr_t)&Blds[bufsel][row * 64 + col];
      asm volatile("global_load_async_to_lds_b128 %0, %1, off"
                   :: "v"(ldsoff), "v"((unsigned long long)(uintptr_t)g)
                   : "memory");
    }
  };

  v8f acc[4];
#pragma unroll
  for (int nt = 0; nt < 4; ++nt)
    acc[nt] = (v8f){0.f, 0.f, 0.f, 0.f, 0.f, 0.f, 0.f, 0.f};

  stage(0, 0);
  int buf = 0;
  for (int kk = 0; kk < K; kk += 64) {
    asm volatile("s_wait_asynccnt 0" ::: "memory");  // own async copies landed
    __syncthreads();                                 // ...visible to all waves
    if (kk + 64 < K) {
      stage(kk + 64, buf ^ 1);
      __builtin_prefetch(Arow + kk + 64, 0, 1);      // global_prefetch_b8 on A stream
    }
#pragma unroll
    for (int kl = 0; kl < 64; kl += 32) {
      const bf16_t* ap = Arow + kk + kl + hl * 8;
      v16bf afr = frag_from2(ap, ap + 16);
#pragma unroll
      for (int nt = 0; nt < 4; ++nt) {
        const bf16_t* bp = &Blds[buf][(nt * 16 + lid) * 64 + kl + hl * 8];
        acc[nt] = wmma_bf16(afr, frag_from2(bp, bp + 16), acc[nt]);
      }
    }
    buf ^= 1;
  }

  float* Cb = Cc + (long)b * strideC;
  int row0 = imS * 128 + wave * 16 + hl * 8;
#pragma unroll
  for (int nt = 0; nt < 4; ++nt) {
    int col = jnS * 64 + nt * 16 + lid;
#pragma unroll
    for (int r = 0; r < 8; ++r)
      Cb[(long)(row0 + r) * Nn + col] = acc[nt][r];
  }
}

// ---------------------------------------------------------------- PRoPE apply kernels

// qp[n,h,t,d] = A^{-T} q : y_i = sum_j Minv[j][i] x_j
__global__ __launch_bounds__(256) void rope_q_kernel(const float* __restrict__ qf,
    const float* __restrict__ Minv, bf16_t* __restrict__ qp) {
  long idx = (long)blockIdx.x * blockDim.x + threadIdx.x;
  if (idx >= (long)kN * kL * kNH * 16) return;
  int g = (int)(idx & 15);
  long r = idx >> 4;
  int h = (int)(r & (kNH - 1)); r >>= 2;
  int t = (int)(r & (kL - 1));  r >>= 10;
  int n = (int)r;
  const float* Mi = Minv + ((long)n * kL + t) * 16;
  const float* xp = qf + ((long)n * kL + t) * kC + h * kHD + g * 4;
  float x0 = xp[0], x1 = xp[1], x2 = xp[2], x3 = xp[3];
  bf16_t* y = qp + (((long)(n * kNH + h) * kL + t) * kHD + g * 4);
#pragma unroll
  for (int i = 0; i < 4; ++i)
    y[i] = (bf16_t)(Mi[0 * 4 + i] * x0 + Mi[1 * 4 + i] * x1 +
                    Mi[2 * 4 + i] * x2 + Mi[3 * 4 + i] * x3);
}

// kp[n,h,t,d] = A k ; vpT[n,h,d,t] = A v
__global__ __launch_bounds__(256) void rope_kv_kernel(const float* __restrict__ kf,
    const float* __restrict__ vf, const float* __restrict__ Mm,
    bf16_t* __restrict__ kp, bf16_t* __restrict__ vpT, int kvLen) {
  long idx = (long)blockIdx.x * blockDim.x + threadIdx.x;
  if (idx >= (long)kN * kvLen * kNH * 16) return;
  int g = (int)(idx & 15);
  long r = idx >> 4;
  int h = (int)(r % kNH); r /= kNH;
  int t = (int)(r % kvLen);
  int n = (int)(r / kvLen);
  const float* M = Mm + ((long)n * kvLen + t) * 16;
  const float* xk = kf + ((long)n * kvLen + t) * kC + h * kHD + g * 4;
  const float* xv = vf + ((long)n * kvLen + t) * kC + h * kHD + g * 4;
  float k0 = xk[0], k1 = xk[1], k2 = xk[2], k3 = xk[3];
  float v0 = xv[0], v1 = xv[1], v2 = xv[2], v3 = xv[3];
  bf16_t* yk = kp + (((long)(n * kNH + h) * kvLen + t) * kHD + g * 4);
  long vb = ((long)(n * kNH + h) * kHD + g * 4) * kvLen + t;
#pragma unroll
  for (int i = 0; i < 4; ++i) {
    yk[i] = (bf16_t)(M[i * 4 + 0] * k0 + M[i * 4 + 1] * k1 +
                     M[i * 4 + 2] * k2 + M[i * 4 + 3] * k3);
    vpT[vb + (long)i * kvLen] =
        (bf16_t)(M[i * 4 + 0] * v0 + M[i * 4 + 1] * v1 +
                 M[i * 4 + 2] * v2 + M[i * 4 + 3] * v3);
  }
}

// o2[n,t,h*64+d] = A^{-1} O : y_i = sum_j Minv[i][j] x_j  (folds head transpose)
__global__ __launch_bounds__(256) void unrope_kernel(const float* __restrict__ O,
    const float* __restrict__ Minv, bf16_t* __restrict__ o2) {
  long idx = (long)blockIdx.x * blockDim.x + threadIdx.x;
  if (idx >= (long)kN * kL * kNH * 16) return;
  int g = (int)(idx & 15);
  long r = idx >> 4;
  int h = (int)(r & (kNH - 1)); r >>= 2;
  int t = (int)(r & (kL - 1));  r >>= 10;
  int n = (int)r;
  const float* Mi = Minv + ((long)n * kL + t) * 16;
  const float* xp = O + (((long)(n * kNH + h) * kL + t) * kHD + g * 4);
  float x0 = xp[0], x1 = xp[1], x2 = xp[2], x3 = xp[3];
  bf16_t* y = o2 + ((long)n * kL + t) * kC + h * kHD + g * 4;
#pragma unroll
  for (int i = 0; i < 4; ++i)
    y[i] = (bf16_t)(Mi[i * 4 + 0] * x0 + Mi[i * 4 + 1] * x1 +
                    Mi[i * 4 + 2] * x2 + Mi[i * 4 + 3] * x3);
}

// ---------------------------------------------------------------- flash attention
// One wave = one 16-row q tile for a given (n, head). Online softmax over 32-wide
// kv chunks. P bounced through per-wave LDS (C-frag -> A-frag), ordered with
// s_wait_dscnt (same-wave DS ops are in-order on CDNA5).

__global__ __launch_bounds__(256) void flash_kernel(const bf16_t* __restrict__ qp,
    const bf16_t* __restrict__ kp, const bf16_t* __restrict__ vpT,
    float* __restrict__ O, int kvLen) {
  __shared__ alignas(16) bf16_t plds[8][16 * 32];
  int lane = threadIdx.x & 31;
  int wave = threadIdx.x >> 5;
  int hl = lane >> 4;
  int lid = lane & 15;
  long task = (long)blockIdx.x * 8 + wave;      // kN*kNH*(kL/16) tasks, exact fill
  int qt = (int)(task % (kL / 16));
  int hn = (int)(task / (kL / 16));             // n*kNH + h
  const bf16_t* qbase = qp + (long)hn * kL * kHD;
  const bf16_t* kbase = kp + (long)hn * kvLen * kHD;
  const bf16_t* vbase = vpT + (long)hn * kHD * kvLen;

  v16bf qa0, qa1;
  {
    const bf16_t* p = qbase + (long)(qt * 16 + lid) * kHD + hl * 8;
    qa0 = frag_from2(p, p + 16);
    qa1 = frag_from2(p + 32, p + 48);
  }
  v8f o0 = {0.f,0.f,0.f,0.f,0.f,0.f,0.f,0.f};
  v8f o1 = o0, o2a = o0, o3 = o0;
  float rmax[8], rsum[8];
#pragma unroll
  for (int r = 0; r < 8; ++r) { rmax[r] = -3.0e38f; rsum[r] = 0.f; }

  for (int j = 0; j < kvLen; j += 32) {
    v8f s0 = {0.f,0.f,0.f,0.f,0.f,0.f,0.f,0.f};
    v8f s1 = s0;
    {
      const bf16_t* p = kbase + (long)(j + lid) * kHD + hl * 8;
      v16bf b0 = frag_from2(p, p + 16);
      v16bf b1 = frag_from2(p + 32, p + 48);
      s0 = wmma_bf16(qa0, b0, s0);
      s0 = wmma_bf16(qa1, b1, s0);
      p = kbase + (long)(j + 16 + lid) * kHD + hl * 8;
      b0 = frag_from2(p, p + 16);
      b1 = frag_from2(p + 32, p + 48);
      s1 = wmma_bf16(qa0, b0, s1);
      s1 = wmma_bf16(qa1, b1, s1);
    }
#pragma unroll
    for (int r = 0; r < 8; ++r) {
      float a0 = s0[r] * 0.125f, a1 = s1[r] * 0.125f;  // 1/sqrt(64)
      float loc = fmaxf(a0, a1);
      loc = fmaxf(loc, __shfl_xor(loc, 1, 32));
      loc = fmaxf(loc, __shfl_xor(loc, 2, 32));
      loc = fmaxf(loc, __shfl_xor(loc, 4, 32));
      loc = fmaxf(loc, __shfl_xor(loc, 8, 32));
      float nm = fmaxf(rmax[r], loc);
      float alpha = __expf(rmax[r] - nm);
      rmax[r] = nm;
      float p0 = __expf(a0 - nm), p1 = __expf(a1 - nm);
      float ps = p0 + p1;
      ps += __shfl_xor(ps, 1, 32);
      ps += __shfl_xor(ps, 2, 32);
      ps += __shfl_xor(ps, 4, 32);
      ps += __shfl_xor(ps, 8, 32);
      rsum[r] = rsum[r] * alpha + ps;
      o0[r] *= alpha; o1[r] *= alpha; o2a[r] *= alpha; o3[r] *= alpha;
      plds[wave][(r + 8 * hl) * 32 + lid] = (bf16_t)p0;
      plds[wave][(r + 8 * hl) * 32 + 16 + lid] = (bf16_t)p1;
    }
    asm volatile("s_wait_dscnt 0" ::: "memory");   // same-wave LDS RAW ordering
    v16bf pa = frag_from2(&plds[wave][lid * 32 + hl * 8],
                          &plds[wave][lid * 32 + hl * 8 + 16]);
    {
      const bf16_t* p = vbase + (long)(0 * 16 + lid) * kvLen + j + hl * 8;
      o0 = wmma_bf16(pa, frag_from2(p, p + 16), o0);
      p = vbase + (long)(1 * 16 + lid) * kvLen + j + hl * 8;
      o1 = wmma_bf16(pa, frag_from2(p, p + 16), o1);
      p = vbase + (long)(2 * 16 + lid) * kvLen + j + hl * 8;
      o2a = wmma_bf16(pa, frag_from2(p, p + 16), o2a);
      p = vbase + (long)(3 * 16 + lid) * kvLen + j + hl * 8;
      o3 = wmma_bf16(pa, frag_from2(p, p + 16), o3);
    }
  }
  float* Ob = O + ((long)hn * kL + qt * 16) * kHD;
#pragma unroll
  for (int r = 0; r < 8; ++r) {
    float inv = 1.0f / rsum[r];
    int ro = r + 8 * hl;
    Ob[(long)ro * kHD + 0 * 16 + lid] = o0[r] * inv;
    Ob[(long)ro * kHD + 1 * 16 + lid] = o1[r] * inv;
    Ob[(long)ro * kHD + 2 * 16 + lid] = o2a[r] * inv;
    Ob[(long)ro * kHD + 3 * 16 + lid] = o3[r] * inv;
  }
}

// ---------------------------------------------------------------- layernorm kernels

__device__ inline float wave_sum(float v) {
  v += __shfl_xor(v, 1, 32);
  v += __shfl_xor(v, 2, 32);
  v += __shfl_xor(v, 4, 32);
  v += __shfl_xor(v, 8, 32);
  v += __shfl_xor(v, 16, 32);
  return v;
}

// x += LN(msg)*w + b ; also refresh bf16 copy of x
__global__ __launch_bounds__(256) void ln_residual_kernel(const float* __restrict__ msg,
    const float* __restrict__ w, const float* __restrict__ bvec,
    float* __restrict__ x, bf16_t* __restrict__ xh) {
  int lane = threadIdx.x & 31;
  int wave = threadIdx.x >> 5;
  long row = (long)blockIdx.x * 8 + wave;        // kN*kL rows, exact fill
  const float* m = msg + row * kC;
  float vals[8];
  float s = 0.f;
#pragma unroll
  for (int j = 0; j < 8; ++j) { vals[j] = m[lane + 32 * j]; s += vals[j]; }
  float mu = wave_sum(s) * (1.0f / kC);
  float vs = 0.f;
#pragma unroll
  for (int j = 0; j < 8; ++j) { float d = vals[j] - mu; vs += d * d; }
  float rs = rsqrtf(wave_sum(vs) * (1.0f / kC) + 1e-5f);
  float* xr = x + row * kC;
  bf16_t* xhr = xh + row * kC;
#pragma unroll
  for (int j = 0; j < 8; ++j) {
    int c = lane + 32 * j;
    float o = xr[c] + (vals[j] - mu) * rs * w[c] + bvec[c];
    xr[c] = o;
    xhr[c] = (bf16_t)o;
  }
}

// cat = [bf16(x), bf16(LN(msg)*w+b)]  (x unchanged)
__global__ __launch_bounds__(256) void ln_cat_kernel(const float* __restrict__ x,
    const float* __restrict__ msg, const float* __restrict__ w,
    const float* __restrict__ bvec, bf16_t* __restrict__ cat) {
  int lane = threadIdx.x & 31;
  int wave = threadIdx.x >> 5;
  long row = (long)blockIdx.x * 8 + wave;
  const float* m = msg + row * kC;
  float vals[8];
  float s = 0.f;
#pragma unroll
  for (int j = 0; j < 8; ++j) { vals[j] = m[lane + 32 * j]; s += vals[j]; }
  float mu = wave_sum(s) * (1.0f / kC);
  float vs = 0.f;
#pragma unroll
  for (int j = 0; j < 8; ++j) { float d = vals[j] - mu; vs += d * d; }
  float rs = rsqrtf(wave_sum(vs) * (1.0f / kC) + 1e-5f);
  const float* xr = x + row * kC;
  bf16_t* cr = cat + row * kFFNin;
#pragma unroll
  for (int j = 0; j < 8; ++j) {
    int c = lane + 32 * j;
    cr[c] = (bf16_t)xr[c];
    cr[kC + c] = (bf16_t)((vals[j] - mu) * rs * w[c] + bvec[c]);
  }
}

__global__ __launch_bounds__(256) void gelu_kernel(const float* __restrict__ in,
                                                   bf16_t* __restrict__ out, long n) {
  long i = (long)blockIdx.x * blockDim.x + threadIdx.x;
  if (i >= n) return;
  float v = in[i];
  out[i] = (bf16_t)(0.5f * v * (1.0f + erff(v * 0.70710678118654752f)));
}

// ---------------------------------------------------------------- host orchestration

extern "C" void kernel_launch(void* const* d_in, const int* in_sizes, int n_in,
                              void* d_out, int out_size, void* d_ws, size_t ws_size,
                              hipStream_t stream) {
  (void)in_sizes; (void)n_in; (void)out_size; (void)ws_size;
  const float* features = (const float*)d_in[0];
  const float* viewmats = (const float*)d_in[1];
  const float* Ks       = (const float*)d_in[2];
  const float* Wq  = (const float*)d_in[3];
  const float* Wk  = (const float*)d_in[4];
  const float* Wv  = (const float*)d_in[5];
  const float* Wm  = (const float*)d_in[6];
  const float* n1w = (const float*)d_in[7];
  const float* n1b = (const float*)d_in[8];
  const float* W1  = (const float*)d_in[9];
  const float* W2  = (const float*)d_in[10];
  const float* n2w = (const float*)d_in[11];
  const float* n2b = (const float*)d_in[12];
  float* out = (float*)d_out;

  char* ws = (char*)d_ws;
  size_t off = 0;
  auto alloc = [&](size_t bytes) -> void* {
    void* p = ws + off;
    off = (off + bytes + 255) & ~(size_t)255;
    return p;
  };

  float*  Mq   = (float*)alloc((size_t)kN * kL * 16 * 4);
  float*  Minv = (float*)alloc((size_t)kN * kL * 16 * 4);
  float*  Mkv  = (float*)alloc((size_t)kN * kKVL * 16 * 4);
  float*  x    = (float*)alloc((size_t)kN * kL * kC * 4);
  bf16_t* xh   = (bf16_t*)alloc((size_t)kN * kL * kC * 2);
  bf16_t* kvxh = (bf16_t*)alloc((size_t)kN * kKVL * kC * 2);
  float*  qf   = (float*)alloc((size_t)kN * kL * kC * 4);
  float*  kf   = (float*)alloc((size_t)kN * kKVL * kC * 4);
  float*  vf   = (float*)alloc((size_t)kN * kKVL * kC * 4);
  bf16_t* qp   = (bf16_t*)alloc((size_t)kN * kNH * kL * kHD * 2);
  bf16_t* kp   = (bf16_t*)alloc((size_t)kN * kNH * kKVL * kHD * 2);
  bf16_t* vpT  = (bf16_t*)alloc((size_t)kN * kNH * kHD * kKVL * 2);
  float*  Of   = (float*)alloc((size_t)kN * kNH * kL * kHD * 4);
  bf16_t* o2   = (bf16_t*)alloc((size_t)kN * kL * kC * 2);
  float*  msg  = (float*)alloc((size_t)kN * kL * kC * 4);
  bf16_t* cat  = (bf16_t*)alloc((size_t)kN * kL * kFFNin * 2);
  float*  hbuf = (float*)alloc((size_t)kN * kL * kFFNhid * 4);
  bf16_t* hh   = (bf16_t*)alloc((size_t)kN * kL * kFFNhid * 2);
  float*  ho   = (float*)alloc((size_t)kN * kL * kC * 4);
  bf16_t* Wqh  = (bf16_t*)alloc((size_t)2 * 2 * kC * kC * 2);
  bf16_t* Wkh  = (bf16_t*)alloc((size_t)2 * 2 * kC * kC * 2);
  bf16_t* Wvh  = (bf16_t*)alloc((size_t)2 * 2 * kC * kC * 2);
  bf16_t* Wmh  = (bf16_t*)alloc((size_t)2 * 2 * kC * kC * 2);
  bf16_t* W1h  = (bf16_t*)alloc((size_t)2 * kFFNhid * kFFNin * 2);
  bf16_t* W2h  = (bf16_t*)alloc((size_t)2 * kC * kFFNhid * 2);

  auto cvt = [&](const float* src, bf16_t* dst, long n) {
    cvt_bf16_kernel<<<cdiv(n, 256), 256, 0, stream>>>(src, dst, n);
  };
  auto gemm = [&](const bf16_t* A, const bf16_t* Bw, float* Cc, int M, int Nn, int K,
                  long sA, long sC) {
    int blocks = kN * (M / 128) * (Nn / 64);   // exact for all shapes used
    gemm_nt_kernel<<<blocks, 256, 0, stream>>>(A, Bw, Cc, M, Nn, K, sA, sC);
  };

  cvt(Wq, Wqh, 2L * 2 * kC * kC);
  cvt(Wk, Wkh, 2L * 2 * kC * kC);
  cvt(Wv, Wvh, 2L * 2 * kC * kC);
  cvt(Wm, Wmh, 2L * 2 * kC * kC);
  cvt(W1, W1h, 2L * kFFNhid * kFFNin);
  cvt(W2, W2h, 2L * kC * kFFNhid);

  build_mq_kernel<<<cdiv((long)kN * kL, 256), 256, 0, stream>>>(viewmats, Ks, Mq, Minv);
  build_mkv_kernel<<<cdiv((long)kN * kKVL, 256), 256, 0, stream>>>(viewmats, Ks, Mkv);
  feat2x_kernel<<<cdiv((long)kN * kL * kC, 256), 256, 0, stream>>>(features, x, xh);

  const long ropeQ = (long)kN * kL * kNH * 16;
  const int flashBlocks = kN * kNH * (kL / 16) / 8;

  for (int i = 0; i < 2; ++i) {
    // kv source captured before this layer's self-attention
    gather_kv_kernel<<<cdiv((long)kN * kKVL * kC, 256), 256, 0, stream>>>(xh, kvxh);

    // ---- self-attention (slot 0), kv cameras == q cameras
    const bf16_t* wq0 = Wqh + (long)(i * 2 + 0) * kC * kC;
    const bf16_t* wk0 = Wkh + (long)(i * 2 + 0) * kC * kC;
    const bf16_t* wv0 = Wvh + (long)(i * 2 + 0) * kC * kC;
    const bf16_t* wm0 = Wmh + (long)(i * 2 + 0) * kC * kC;
    gemm(xh, wq0, qf, kL, kC, kC, (long)kL * kC, (long)kL * kC);
    gemm(xh, wk0, kf, kL, kC, kC, (long)kL * kC, (long)kL * kC);
    gemm(xh, wv0, vf, kL, kC, kC, (long)kL * kC, (long)kL * kC);
    rope_q_kernel<<<cdiv(ropeQ, 256), 256, 0, stream>>>(qf, Minv, qp);
    rope_kv_kernel<<<cdiv((long)kN * kL * kNH * 16, 256), 256, 0, stream>>>(
        kf, vf, Mq, kp, vpT, kL);
    flash_kernel<<<flashBlocks, 256, 0, stream>>>(qp, kp, vpT, Of, kL);
    unrope_kernel<<<cdiv(ropeQ, 256), 256, 0, stream>>>(Of, Minv, o2);
    gemm(o2, wm0, msg, kL, kC, kC, (long)kL * kC, (long)kL * kC);
    ln_residual_kernel<<<kN * kL / 8, 256, 0, stream>>>(
        msg, n1w + (long)(i * 2 + 0) * kC, n1b + (long)(i * 2 + 0) * kC, x, xh);

    // ---- cross-attention (slot 1) + FFN
    const bf16_t* wq1 = Wqh + (long)(i * 2 + 1) * kC * kC;
    const bf16_t* wk1 = Wkh + (long)(i * 2 + 1) * kC * kC;
    const bf16_t* wv1 = Wvh + (long)(i * 2 + 1) * kC * kC;
    const bf16_t* wm1 = Wmh + (long)(i * 2 + 1) * kC * kC;
    gemm(xh, wq1, qf, kL, kC, kC, (long)kL * kC, (long)kL * kC);
    gemm(kvxh, wk1, kf, kKVL, kC, kC, (long)kKVL * kC, (long)kKVL * kC);
    gemm(kvxh, wv1, vf, kKVL, kC, kC, (long)kKVL * kC, (long)kKVL * kC);
    rope_q_kernel<<<cdiv(ropeQ, 256), 256, 0, stream>>>(qf, Minv, qp);
    rope_kv_kernel<<<cdiv((long)kN * kKVL * kNH * 16, 256), 256, 0, stream>>>(
        kf, vf, Mkv, kp, vpT, kKVL);
    flash_kernel<<<flashBlocks, 256, 0, stream>>>(qp, kp, vpT, Of, kKVL);
    unrope_kernel<<<cdiv(ropeQ, 256), 256, 0, stream>>>(Of, Minv, o2);
    gemm(o2, wm1, msg, kL, kC, kC, (long)kL * kC, (long)kL * kC);
    ln_cat_kernel<<<kN * kL / 8, 256, 0, stream>>>(
        x, msg, n1w + (long)(i * 2 + 1) * kC, n1b + (long)(i * 2 + 1) * kC, cat);
    gemm(cat, W1h + (long)i * kFFNhid * kFFNin, hbuf, kL, kFFNhid, kFFNin,
         (long)kL * kFFNin, (long)kL * kFFNhid);
    gelu_kernel<<<cdiv((long)kN * kL * kFFNhid, 256), 256, 0, stream>>>(
        hbuf, hh, (long)kN * kL * kFFNhid);
    gemm(hh, W2h + (long)i * kC * kFFNhid, ho, kL, kC, kFFNhid,
         (long)kL * kFFNhid, (long)kL * kC);
    ln_residual_kernel<<<kN * kL / 8, 256, 0, stream>>>(
        ho, n2w + (long)i * kC, n2b + (long)i * kC, x, xh);
  }

  out_kernel<<<cdiv((long)kN * kC * kL, 256), 256, 0, stream>>>(x, out);
}
